// Net_8263517078029
// MI455X (gfx1250) — matrix operand order repaired and checked by reading, not compiled, not run on hardware
//
#include <hip/hip_runtime.h>

#define N_NODES 50000
#define N_EDGES 500000
#define INC 100
#define HIDN 80
#define EDGED 100
#define AVG_LOG_C 0.6839635f
#define AVG_LIN_C 1.4712042f

typedef __attribute__((ext_vector_type(16))) _Float16 v16h;
typedef __attribute__((ext_vector_type(4)))  _Float16 v4h;
typedef __attribute__((ext_vector_type(8)))  float    v8f;

#define BM 64
#define BN 64
#define BK 32
#define LDA 36   /* halfs per LDS row: 72 bytes -> 8B aligned rows, no 4-way bank conflict */

// ------------- Generic C = act(A @ B + bias), fp32 in/out, f16 WMMA core ------
// Requires: K % 4 == 0, N % 4 == 0 (true for every call site here).
__global__ __launch_bounds__(256)
void gemm_wmma_f16(const float* __restrict__ A, const float* __restrict__ B,
                   const float* __restrict__ bias, float* __restrict__ C,
                   int M, int N, int K, int relu)
{
  __shared__ _Float16 As[BM][LDA];   // [row][k]
  __shared__ _Float16 BsT[BN][LDA];  // [col][k] (transposed stage)
  const int tid  = threadIdx.x;
  const int lane = tid & 31;
  const int wave = tid >> 5;         // 8 waves
  const int wm   = wave & 3;         // 4 M-tiles of 16
  const int wn   = wave >> 2;        // 2 N-strips of 32
  const int m0   = blockIdx.y * BM;
  const int n0   = blockIdx.x * BN;
  const int hl   = lane >> 4;        // half-wave select
  const int l15  = lane & 15;

  v8f acc0 = {}, acc1 = {};
  const int nsteps = (K + BK - 1) / BK;
  for (int ks = 0; ks < nsteps; ++ks) {
    const int k0 = ks * BK;
    // ---- stage A: BMxBK via branchless float4 loads (clamped index, zero-select)
#pragma unroll
    for (int i = 0; i < (BM * BK) / (256 * 4); ++i) {
      int g  = i * 256 + tid;            // float4 group
      int r  = g >> 3;                   // 8 groups per 32-wide row
      int kq = (g & 7) * 4;
      int gr = m0 + r, gk = k0 + kq;
      bool ok = (gr < M) && (gk + 3 < K);
      int grc = min(gr, M - 1);
      int gkc = min(gk, K - 4);          // stays %4==0 -> 16B aligned
      float4 v = *(const float4*)(A + (long long)grc * K + gkc);
      if (!ok) { v.x = 0.f; v.y = 0.f; v.z = 0.f; v.w = 0.f; }
      v4h h; h[0] = (_Float16)v.x; h[1] = (_Float16)v.y;
             h[2] = (_Float16)v.z; h[3] = (_Float16)v.w;
      *(v4h*)&As[r][kq] = h;
    }
    // ---- stage B: BKxBN via branchless float4 loads, transposed b16 stores
#pragma unroll
    for (int i = 0; i < (BK * BN) / (256 * 4); ++i) {
      int g  = i * 256 + tid;
      int k  = g >> 4;                   // 16 groups per 64-wide row
      int nq = (g & 15) * 4;
      int gk = k0 + k, gn = n0 + nq;
      bool ok = (gk < K) && (gn + 3 < N);
      int gkc = min(gk, K - 1);
      int gnc = min(gn, N - 4);          // stays %4==0 -> 16B aligned
      float4 v = *(const float4*)(B + (long long)gkc * N + gnc);
      if (!ok) { v.x = 0.f; v.y = 0.f; v.z = 0.f; v.w = 0.f; }
      BsT[nq + 0][k] = (_Float16)v.x;
      BsT[nq + 1][k] = (_Float16)v.y;
      BsT[nq + 2][k] = (_Float16)v.z;
      BsT[nq + 3][k] = (_Float16)v.w;
    }
    __syncthreads();
    // ---- fragments: A and B both use contiguous k = 16*hl + s (same order)
    v16h af, bf0, bf1;
#pragma unroll
    for (int j = 0; j < 8; ++j) {
      int kb = 2 * j + 16 * hl;
      af[2 * j]     = As[wm * 16 + l15][kb];
      af[2 * j + 1] = As[wm * 16 + l15][kb + 1];
    }
#pragma unroll
    for (int t = 0; t < 16; ++t) {
      bf0[t] = BsT[wn * 32 + l15][t + 16 * hl];
      bf1[t] = BsT[wn * 32 + 16 + l15][t + 16 * hl];
    }
    acc0 = __builtin_amdgcn_wmma_f32_16x16x32_f16(false, af, false, bf0,
                                                  (short)0, acc0, false, false);
    acc1 = __builtin_amdgcn_wmma_f32_16x16x32_f16(false, af, false, bf1,
                                                  (short)0, acc1, false, false);
    __syncthreads();
  }
  // ---- epilogue
#pragma unroll
  for (int u = 0; u < 2; ++u) {
    int col = n0 + wn * 32 + u * 16 + l15;
    float bv = (bias && col < N) ? bias[col] : 0.f;
    const v8f& acc = u ? acc1 : acc0;
#pragma unroll
    for (int r = 0; r < 8; ++r) {
      int row = m0 + wm * 16 + r + 8 * hl;
      if (row < M && col < N) {
        float v = acc[r] + bv;
        if (relu) v = fmaxf(v, 0.f);
        C[(long long)row * N + col] = v;
      }
    }
  }
}

// ---------------- float atomic min/max via monotone int mapping ---------------
__device__ inline void atomicMaxF(float* a, float v) {
  if (v >= 0.f) atomicMax((int*)a, __float_as_int(v));
  else          atomicMin((unsigned int*)a, __float_as_uint(v));
}
__device__ inline void atomicMinF(float* a, float v) {
  if (v >= 0.f) atomicMin((int*)a, __float_as_int(v));
  else          atomicMax((unsigned int*)a, __float_as_uint(v));
}

// ---------------- small helper kernels ---------------------------------------
__global__ void init_aggr(float* sum, float* sumsq, float* mn, float* mx,
                          float* cnt, int NF) {
  int i = blockIdx.x * blockDim.x + threadIdx.x;
  if (i < NF) {
    sum[i] = 0.f; sumsq[i] = 0.f;
    mn[i] = __int_as_float(0x7F800000);   // +inf
    mx[i] = __int_as_float(0xFF800000);   // -inf
  }
  if (i < N_NODES) cnt[i] = 0.f;
}

__global__ void edge_count(const int* __restrict__ dst, float* cnt) {
  int e = blockIdx.x * blockDim.x + threadIdx.x;
  if (e < N_EDGES) atomicAdd(&cnt[dst[e]], 1.0f);
}

__global__ void edge_scatter(const float* __restrict__ eh,
                             const float* __restrict__ xi,
                             const float* __restrict__ xj,
                             const int* __restrict__ src,
                             const int* __restrict__ dst,
                             float* sum, float* sumsq, float* mn, float* mx,
                             int F) {
  int i = blockIdx.x * blockDim.x + threadIdx.x;
  int tot = N_EDGES * F;
  if (i >= tot) return;
  int e = i / F, f = i - e * F;
  int d = dst[e], s = src[e];
  float h = eh[i] + xi[d * F + f] + xj[s * F + f];
  int di = d * F + f;
  atomicAdd(&sum[di], h);
  atomicAdd(&sumsq[di], h * h);
  atomicMaxF(&mx[di], h);
  atomicMinF(&mn[di], h);
}

__global__ void node_post(const float* __restrict__ sum,
                          const float* __restrict__ sumsq,
                          const float* __restrict__ mn,
                          const float* __restrict__ mx,
                          const float* __restrict__ cnt,
                          float* __restrict__ agg, int F) {
  int i = blockIdx.x * blockDim.x + threadIdx.x;
  int tot = N_NODES * F;
  if (i >= tot) return;
  int n = i / F, f = i - n * F;
  float c = cnt[n];
  float deg = fmaxf(c, 1.f);
  float s = sum[i];
  float mean = s / deg;
  float msq = sumsq[i] / deg;
  float var = msq - mean * mean;
  float sd = sqrtf(fmaxf(var, 0.f) + 1e-5f);
  bool has = c > 0.f;
  float mnv = has ? mn[i] : 0.f;
  float mxv = has ? mx[i] : 0.f;
  float* a = agg + (long long)n * 6 * F;
  a[f] = s; a[F + f] = mean; a[2 * F + f] = mnv;
  a[3 * F + f] = mxv; a[4 * F + f] = var; a[5 * F + f] = sd;
}

__global__ void make_scales(const float* __restrict__ cnt, float* __restrict__ scal) {
  int n = blockIdx.x * blockDim.x + threadIdx.x;
  if (n >= N_NODES) return;
  float deg = fmaxf(cnt[n], 1.f);
  float logd = logf(deg + 1.f);
  scal[4 * n + 0] = logd / AVG_LOG_C;
  scal[4 * n + 1] = AVG_LOG_C / logd;
  scal[4 * n + 2] = deg / AVG_LIN_C;
  scal[4 * n + 3] = AVG_LIN_C / deg;
}

// out = outpre + bpost + Z0 + s1*Z1 + s2*Z2 + s3*Z3 + s4*Z4   (Z is N x 5H)
__global__ void combine_post(const float* __restrict__ outpre,
                             const float* __restrict__ Z,
                             const float* __restrict__ scal,
                             const float* __restrict__ bpost,
                             float* __restrict__ outb, int H) {
  int i = blockIdx.x * blockDim.x + threadIdx.x;
  int tot = N_NODES * H;
  if (i >= tot) return;
  int n = i / H, j = i - n * H;
  const float* z = Z + (long long)n * 5 * H;
  float v = outpre[i] + bpost[j] + z[j]
          + scal[4 * n + 0] * z[H + j]
          + scal[4 * n + 1] * z[2 * H + j]
          + scal[4 * n + 2] * z[3 * H + j]
          + scal[4 * n + 3] * z[4 * H + j];
  outb[i] = v;
}

// Bcat[r][k*H+j] = Wpost[(F + k*6F + r)*H + j], r<6F, k<5, j<H
__global__ void repack_bcat(const float* __restrict__ Wpost,
                            float* __restrict__ Bcat, int F) {
  const int H = HIDN;
  int tot = 6 * F * 5 * H;
  int i = blockIdx.x * blockDim.x + threadIdx.x;
  if (i >= tot) return;
  int r = i / (5 * H);
  int c = i - r * (5 * H);
  int k = c / H, j = c - k * H;
  Bcat[i] = Wpost[(F + k * 6 * F + r) * H + j];
}

// bc[j] = sum_i be[i]*We_part[i,j] + bpre[j]
__global__ void make_bc(const float* __restrict__ be, const float* __restrict__ Wep,
                        const float* __restrict__ bpre, float* __restrict__ bc,
                        int Kd, int F) {
  int j = blockIdx.x * blockDim.x + threadIdx.x;
  if (j >= F) return;
  float acc = bpre[j];
  for (int i = 0; i < Kd; ++i) acc += be[i] * Wep[i * F + j];
  bc[j] = acc;
}

// ---------------- host orchestration -----------------------------------------
static inline int cdiv_i(int a, int b) { return (a + b - 1) / b; }

static inline void gemm(hipStream_t s, const float* A, const float* B,
                        const float* bias, float* C, int M, int N, int K, int relu) {
  dim3 g(cdiv_i(N, BN), cdiv_i(M, BM));
  gemm_wmma_f16<<<g, 256, 0, s>>>(A, B, bias, C, M, N, K, relu);
}

extern "C" void kernel_launch(void* const* d_in, const int* in_sizes, int n_in,
                              void* d_out, int out_size, void* d_ws, size_t ws_size,
                              hipStream_t stream) {
  const float* x1  = (const float*)d_in[0];
  const int*   ei1 = (const int*)d_in[1];
  const float* ea1 = (const float*)d_in[2];
  const float* x2  = (const float*)d_in[3];
  const int*   ei2 = (const int*)d_in[4];
  const float* ea2 = (const float*)d_in[5];
  const float *We1 = (const float*)d_in[6],  *be1  = (const float*)d_in[7];
  const float *Wpre1 = (const float*)d_in[8], *bpre1 = (const float*)d_in[9];
  const float *Wpost1 = (const float*)d_in[10], *bpost1 = (const float*)d_in[11];
  const float *Wlin1 = (const float*)d_in[12], *blin1 = (const float*)d_in[13];
  const float *We2 = (const float*)d_in[14], *be2 = (const float*)d_in[15];
  const float *Wpre2 = (const float*)d_in[16], *bpre2 = (const float*)d_in[17];
  const float *Wpost2 = (const float*)d_in[18], *bpost2 = (const float*)d_in[19];
  const float *Wlin2 = (const float*)d_in[20], *blin2 = (const float*)d_in[21];
  const float *Wfc1 = (const float*)d_in[22], *bfc1 = (const float*)d_in[23];
  const float *Wfc2 = (const float*)d_in[24], *bfc2 = (const float*)d_in[25];
  float* out = (float*)d_out;

  // workspace carve-up (floats)
  float* W = (float*)d_ws;
  size_t o = 0;
  auto alloc = [&](size_t n) { float* p = W + o; o += n; return p; };
  float* eh     = alloc((size_t)N_EDGES * INC);    // 50M (also fits E*80)
  float* xi     = alloc((size_t)N_NODES * INC);
  float* xj     = alloc((size_t)N_NODES * INC);
  float* sum    = alloc((size_t)N_NODES * INC);
  float* sumsq  = alloc((size_t)N_NODES * INC);
  float* mn     = alloc((size_t)N_NODES * INC);
  float* mx     = alloc((size_t)N_NODES * INC);
  float* cnt    = alloc(50048);
  float* agg    = alloc((size_t)N_NODES * 6 * INC); // 30M (fits 6*80 too)
  float* scal   = alloc((size_t)N_NODES * 4);
  float* Z      = alloc((size_t)N_NODES * 5 * HIDN);
  float* outpre = alloc((size_t)N_NODES * HIDN);
  float* pbuf   = alloc((size_t)N_NODES * HIDN);
  float* y1     = alloc((size_t)N_NODES * HIDN);
  float* y2     = alloc((size_t)N_NODES * HIDN);
  float* tbuf   = alloc((size_t)N_NODES * HIDN);
  float* Wc1    = alloc(EDGED * INC);
  float* bc1    = alloc(128);
  float* Wc2    = alloc(EDGED * HIDN);
  float* bc2    = alloc(128);
  float* Bcat1  = alloc((size_t)6 * INC * 5 * HIDN);
  float* Bcat2  = alloc((size_t)6 * HIDN * 5 * HIDN);

  const int TB = 256;

  // ---- precompute combined edge weights & repacked post weights (shared) ----
  gemm(stream, We1, Wpre1 + 2 * INC * INC, nullptr, Wc1, EDGED, INC, INC, 0);
  make_bc<<<cdiv_i(INC, TB), TB, 0, stream>>>(be1, Wpre1 + 2 * INC * INC, bpre1, bc1, INC, INC);
  gemm(stream, We2, Wpre2 + 2 * HIDN * HIDN, nullptr, Wc2, EDGED, HIDN, HIDN, 0);
  make_bc<<<cdiv_i(HIDN, TB), TB, 0, stream>>>(be2, Wpre2 + 2 * HIDN * HIDN, bpre2, bc2, HIDN, HIDN);
  repack_bcat<<<cdiv_i(6 * INC * 5 * HIDN, TB), TB, 0, stream>>>(Wpost1, Bcat1, INC);
  repack_bcat<<<cdiv_i(6 * HIDN * 5 * HIDN, TB), TB, 0, stream>>>(Wpost2, Bcat2, HIDN);

  auto pna = [&](const float* xin, int Fin, const int* src, const int* dst,
                 const float* ea, const float* Wc, const float* bc,
                 const float* Wpre, const float* Wpost, const float* bpost,
                 const float* Bcat, const float* Wlin, const float* blin,
                 float* yout, int relu_out) {
    // node-level pre-projections
    gemm(stream, xin, Wpre,             nullptr, xi, N_NODES, Fin, Fin, 0);
    gemm(stream, xin, Wpre + Fin * Fin, nullptr, xj, N_NODES, Fin, Fin, 0);
    // edge-level combined projection (edge_attr @ (We@W_e) + (be@W_e + bpre))
    gemm(stream, ea, Wc, bc, eh, N_EDGES, Fin, EDGED, 0);
    // aggregate
    init_aggr<<<cdiv_i(N_NODES * Fin, TB), TB, 0, stream>>>(sum, sumsq, mn, mx, cnt, N_NODES * Fin);
    edge_count<<<cdiv_i(N_EDGES, TB), TB, 0, stream>>>(dst, cnt);
    edge_scatter<<<cdiv_i(N_EDGES * Fin, TB), TB, 0, stream>>>(eh, xi, xj, src, dst,
                                                               sum, sumsq, mn, mx, Fin);
    node_post<<<cdiv_i(N_NODES * Fin, TB), TB, 0, stream>>>(sum, sumsq, mn, mx, cnt, agg, Fin);
    make_scales<<<cdiv_i(N_NODES, TB), TB, 0, stream>>>(cnt, scal);
    // factored post projection
    gemm(stream, xin, Wpost, nullptr, outpre, N_NODES, HIDN, Fin, 0);           // x @ Wp_x
    gemm(stream, agg, Bcat, nullptr, Z, N_NODES, 5 * HIDN, 6 * Fin, 0);         // agg @ [5 blocks]
    combine_post<<<cdiv_i(N_NODES * HIDN, TB), TB, 0, stream>>>(outpre, Z, scal, bpost, pbuf, HIDN);
    // final linear (+ optional relu)
    gemm(stream, pbuf, Wlin, blin, yout, N_NODES, HIDN, HIDN, relu_out);
  };

  auto branch = [&](const float* x, const int* ei, const float* ea, float* outp) {
    const int* src = ei;
    const int* dst = ei + N_EDGES;
    pna(x,  INC,  src, dst, ea, Wc1, bc1, Wpre1, Wpost1, bpost1, Bcat1, Wlin1, blin1, y1, 1);
    pna(y1, HIDN, src, dst, ea, Wc2, bc2, Wpre2, Wpost2, bpost2, Bcat2, Wlin2, blin2, y2, 0);
    gemm(stream, y2, Wfc1, bfc1, tbuf, N_NODES, HIDN, HIDN, 1);
    gemm(stream, tbuf, Wfc2, bfc2, outp, N_NODES, HIDN, HIDN, 0);
  };

  branch(x1, ei1, ea1, out);
  branch(x2, ei2, ea2, out + (size_t)N_NODES * HIDN);

  (void)in_sizes; (void)n_in; (void)out_size; (void)ws_size;
}
